// HybridMaskedAttention_25838523253123
// MI455X (gfx1250) — compile-verified, matrix operand order checked by reading, not run on hardware
//
#include <hip/hip_runtime.h>

// MI455X / gfx1250, wave32, WMMA f32_16x16x32_f16 path.
// HybridMaskedAttention: shared QKV projections + 3-level hierarchical attention
// + sliding-window attention, output projection folded into one final GEMM.
// Round 4: fix async builtin arg types (wants v4i* in AS1/AS3, per hipcc
// diagnostic). Otherwise identical to round 3: transposed V staging, async
// global->LDS staging of K chunks and pre-transposed f16 weights (ASYNCcnt,
// double-buffered), packed b128 LDS stores.

typedef __attribute__((ext_vector_type(16))) _Float16 v16h;
typedef __attribute__((ext_vector_type(8)))  _Float16 v8h;
typedef __attribute__((ext_vector_type(8)))  float    v8f;
typedef __attribute__((ext_vector_type(4)))  int      v4i;

#define E_DIM 1024
#define HEADS 16
#define HD    64

#if defined(__gfx1250__) &&                                         \
    __has_builtin(__builtin_amdgcn_global_load_async_to_lds_b128) && \
    __has_builtin(__builtin_amdgcn_s_wait_asynccnt)
#define USE_ASYNC_LDS 1
#else
#define USE_ASYNC_LDS 0
#endif

// 16-byte global->LDS stage: async (ASYNCcnt) when available, sync fallback.
__device__ __forceinline__ void stage_b128(const _Float16* g, _Float16* l) {
#if USE_ASYNC_LDS
  typedef __attribute__((address_space(1))) v4i gv4i;
  typedef __attribute__((address_space(3))) v4i lv4i;
  __builtin_amdgcn_global_load_async_to_lds_b128((gv4i*)g, (lv4i*)l, 0, 0);
#else
  *(uint4*)l = *(const uint4*)g;
#endif
}

__device__ __forceinline__ void wait_stage() {
#if USE_ASYNC_LDS
  __builtin_amdgcn_s_wait_asynccnt(0);
#endif
}

__device__ __forceinline__ v8f zero8() {
  v8f z = {0.f, 0.f, 0.f, 0.f, 0.f, 0.f, 0.f, 0.f};
  return z;
}

__device__ __forceinline__ v8f wmma_f16(v16h a, v16h b, v8f c) {
  return __builtin_amdgcn_wmma_f32_16x16x32_f16(false, a, false, b, (short)0, c,
                                                false, false);
}

// A-fragment (16x32 f16) gathered from a row-major matrix (ld in halves).
// Lane l: row = l&15, K-half = l>>4 ; VGPRs 0-3 hold K = ko..ko+7,
// VGPRs 4-7 hold K = 16+ko..16+ko+7  (ISA 16-bit A layout).
// Also used for B-fragments when the source is stored N-major (B^T trick).
// Contiguous per lane -> lowers to 2x ds_load_b128 / global_load_b128.
__device__ __forceinline__ v16h frag_row16(const _Float16* base, int ld) {
  const int l = threadIdx.x & 31;
  const _Float16* p = base + (l & 15) * ld + ((l >> 4) * 8);
  v16h f;
#pragma unroll
  for (int j = 0; j < 8; ++j) {
    f[j]     = p[j];
    f[8 + j] = p[16 + j];
  }
  return f;
}

// ---------------------------------------------------------------------------
// W (f32, KxN row-major) -> Wt (f16, NxK row-major), 32x32 LDS-tiled.
// ---------------------------------------------------------------------------
__global__ __launch_bounds__(256) void transpose_w_kernel(
    const float* __restrict__ in, _Float16* __restrict__ out, int N, int K) {
  __shared__ float tile[32][33];
  const int bn = blockIdx.x * 32;  // n tile
  const int bk = blockIdx.y * 32;  // k tile
  const int tx = threadIdx.x & 31, ty = threadIdx.x >> 5;  // 32 x 8
#pragma unroll
  for (int i = 0; i < 32; i += 8)
    tile[ty + i][tx] = in[(size_t)(bk + ty + i) * N + bn + tx];
  __syncthreads();
#pragma unroll
  for (int i = 0; i < 32; i += 8)
    out[(size_t)(bn + ty + i) * K + bk + tx] = (_Float16)tile[tx][ty + i];
}

// ---------------------------------------------------------------------------
// Tiled WMMA GEMM: C = A(f32, MxK) @ W + biasScale*bias, with W given
// pre-transposed f16 Wt[N][K]. FINAL=0: f16 out. FINAL=1: sigmoid(soft_mask)
// gate, f32 out. Block: 256 threads = 8 waves (4Mx2N); tile 128x128; K-step 32.
// A double-buffered in registers (f32->f16 convert); B async-staged to LDS
// (double-buffered, ASYNCcnt).
// ---------------------------------------------------------------------------
template <int FINAL>
__global__ __launch_bounds__(256) void wmma_gemm_kernel(
    const float* __restrict__ A, const _Float16* __restrict__ Wt,
    const float* __restrict__ bias, float biasScale,
    _Float16* __restrict__ outH, float* __restrict__ outF,
    const float* __restrict__ smask, int M, int N, int K, int Lmod) {
  __shared__ __align__(16) _Float16 As[128 * 32];      // row-major 128 x 32
  __shared__ __align__(16) _Float16 Bst[2][128 * 32];  // [n][k], double-buffered
  const int t = threadIdx.x;
  const int tileN = blockIdx.x * 128;
  const int tileM = blockIdx.y * 128;
  const int wave = t >> 5;
  const int wm = wave & 3;   // 32-row strip
  const int wn = wave >> 2;  // 64-col strip

  v8f acc[2][4];
#pragma unroll
  for (int i = 0; i < 2; ++i)
#pragma unroll
    for (int j = 0; j < 4; ++j) acc[i][j] = zero8();

  float4 rA[4];
  auto load_A = [&](int k0) {
#pragma unroll
    for (int i = 0; i < 4; ++i) {
      const int idx = (t + i * 256) * 4;  // 0..4092 step 4, disjoint
      const int r = idx >> 5, c = idx & 31;
      rA[i] = *(const float4*)(A + (size_t)(tileM + r) * K + (k0 + c));
    }
  };
  // B tile: each thread copies 16 contiguous f16 (two b128) of one Wt row.
  const int bn = t >> 1, bko = (t & 1) * 16;
  auto issue_B = [&](int k0, int buf) {
    const _Float16* src = Wt + (size_t)(tileN + bn) * K + k0 + bko;
    _Float16* dst = &Bst[buf][bn * 32 + bko];
    stage_b128(src, dst);
    stage_b128(src + 8, dst + 8);
  };

  load_A(0);
  issue_B(0, 0);
  int p = 0;
  for (int k0 = 0; k0 < K; k0 += 32) {
    __syncthreads();  // As readers (prev iter) done; Bst[p] writers were pre-wait
#pragma unroll
    for (int i = 0; i < 4; ++i) {
      const int idx = (t + i * 256) * 4;
      const int r = idx >> 5, c = idx & 31;
      As[r * 32 + c + 0] = (_Float16)rA[i].x;
      As[r * 32 + c + 1] = (_Float16)rA[i].y;
      As[r * 32 + c + 2] = (_Float16)rA[i].z;
      As[r * 32 + c + 3] = (_Float16)rA[i].w;
    }
    wait_stage();     // my async copies into Bst[p] landed
    __syncthreads();  // everyone's As + Bst[p] published
    if (k0 + 32 < K) {
      load_A(k0 + 32);        // overlaps WMMA burst below
      issue_B(k0 + 32, p ^ 1);
    }

    const v16h a0 = frag_row16(As + (wm * 32) * 32, 32);
    const v16h a1 = frag_row16(As + (wm * 32 + 16) * 32, 32);
#pragma unroll
    for (int nt = 0; nt < 4; ++nt) {
      const v16h b = frag_row16(&Bst[p][(wn * 64 + nt * 16) * 32], 32);
      acc[0][nt] = wmma_f16(a0, b, acc[0][nt]);
      acc[1][nt] = wmma_f16(a1, b, acc[1][nt]);
    }
    p ^= 1;
  }

  // Epilogue. C layout: lane = N + 16*(M>=8), vgpr j -> row j (+8 for hi half).
  const int l = t & 31;
  const int lane_n = l & 15, hh = l >> 4;
#pragma unroll
  for (int mi = 0; mi < 2; ++mi)
#pragma unroll
    for (int nt = 0; nt < 4; ++nt)
#pragma unroll
      for (int j = 0; j < 8; ++j) {
        const int m = tileM + wm * 32 + mi * 16 + j + 8 * hh;
        const int n = tileN + wn * 64 + nt * 16 + lane_n;
        float v = acc[mi][nt][j] + bias[n] * biasScale;
        if (FINAL) {
          const float sm = smask[m % Lmod];
          const float g = 1.0f / (1.0f + __expf(-sm));
          outF[(size_t)m * N + n] = v * g;
        } else {
          outH[(size_t)m * N + n] = (_Float16)v;
        }
      }
}

// ---------------------------------------------------------------------------
// out[b, l, :] = 0.5*(in[b, 2l, :] + in[b, 2l+1, :])   (f16)
// ---------------------------------------------------------------------------
__global__ void downsample2_kernel(const _Float16* __restrict__ in,
                                   _Float16* __restrict__ out, int Lout) {
  const size_t i = (size_t)blockIdx.x * blockDim.x + threadIdx.x;
  const size_t total = (size_t)32 * Lout * E_DIM;
  if (i >= total) return;
  const int e = (int)(i % E_DIM);
  const size_t r = i / E_DIM;
  const int lo = (int)(r % Lout);
  const int b = (int)(r / Lout);
  const size_t i0 = (((size_t)b * (2 * Lout)) + 2 * lo) * E_DIM + e;
  out[i] = (_Float16)(0.5f * ((float)in[i0] + (float)in[i0 + E_DIM]));
}

// ---------------------------------------------------------------------------
// Flash attention over one (batch, head, 128-query tile); 8 waves x 16 rows.
// Transposed scores: S^T = K_chunk x Q^T (lane owns a query column; softmax is
// in-register VALU + two shfl_xor(16) per chunk). O^T = V^T x P^T with V staged
// TRANSPOSED in LDS so every fragment is a contiguous b128 load. K chunks
// async-staged (ASYNCcnt), double-buffered; V staged d-major with one packed
// b128 store per thread. O^T transposed back through LDS for coalesced f32
// accumulation into ctx.
// ---------------------------------------------------------------------------
__global__ __launch_bounds__(256) void flash_attn_kernel(
    const _Float16* __restrict__ Q, const _Float16* __restrict__ Kt,
    const _Float16* __restrict__ Vt, float* __restrict__ ctx, int Lq, int Lk,
    float outw, int accumulate) {
  __shared__ __align__(16) _Float16 Ks[2][32 * 64];   // [key][d]
  __shared__ __align__(16) _Float16 Vts[2][64 * 32];  // [d][key] (transposed)
  __shared__ __align__(16) _Float16 Pl[8 * 16 * 32];  // per-wave P tiles
  __shared__ __align__(16) float    Ot[8 * 16 * 64];  // 32KB transpose buffer
  const int t = threadIdx.x;
  const int wave = t >> 5;
  const int l = t & 31;
  const int lane_q = l & 15, hh = l >> 4;
  const int batch = blockIdx.x / HEADS;
  const int h = blockIdx.x % HEADS;
  const int qrow0 = blockIdx.y * 128 + wave * 16;

  const _Float16* Qb = Q + ((size_t)batch * Lq + qrow0) * E_DIM + h * HD;
  const v16h qb0 = frag_row16(Qb, E_DIM);       // B-frag of Q^T, d = 0..31
  const v16h qb1 = frag_row16(Qb + 32, E_DIM);  // d = 32..63
  _Float16* Pw = Pl + wave * (16 * 32);
  float* Ow = Ot + wave * (16 * 64);

  float mrun = -1e30f, lrun = 0.f;  // per-lane = per-query stats
  v8f O[4];
#pragma unroll
  for (int nt = 0; nt < 4; ++nt) O[nt] = zero8();

  // K staging: one b128 per thread (row-major). V staging: d-major scalar
  // loads (coalesced across lanes) packed into one b128 LDS store -> Vts.
  const int krow = t >> 3, kcol = (t & 7) * 8;
  const int vdl = t & 63, vks = (t >> 6) * 8;
  auto issue_K = [&](int kc, int buf) {
    const size_t g = ((size_t)batch * Lk + kc + krow) * E_DIM + h * HD + kcol;
    stage_b128(Kt + g, &Ks[buf][krow * 64 + kcol]);
  };
  auto load_V = [&](int kc) {
    v8h r;
#pragma unroll
    for (int j = 0; j < 8; ++j)
      r[j] = Vt[((size_t)batch * Lk + kc + vks + j) * E_DIM + h * HD + vdl];
    return r;
  };

  issue_K(0, 0);
  {
    const v8h rv = load_V(0);
    *(v8h*)(&Vts[0][vdl * 32 + vks]) = rv;
  }
  wait_stage();
  __syncthreads();

  const float scale = 0.125f;  // 1/sqrt(64)
  int p = 0;
  for (int kc = 0; kc < Lk; kc += 32, p ^= 1) {
    const bool pref = (kc + 32) < Lk;
    v8h rv;
    if (pref) {
      issue_K(kc + 32, p ^ 1);  // async, lands during compute
      rv = load_V(kc + 32);     // in flight during compute
    }
    const _Float16* Kc = Ks[p];
    const _Float16* Vc = Vts[p];

    // S^T (keys x queries): A = K rows (contiguous d), B = Q^T via B^T trick
    v8f s0 = wmma_f16(frag_row16(Kc, 64), qb0, zero8());
    s0 = wmma_f16(frag_row16(Kc + 32, 64), qb1, s0);
    v8f s1 = wmma_f16(frag_row16(Kc + 16 * 64, 64), qb0, zero8());
    s1 = wmma_f16(frag_row16(Kc + 16 * 64 + 32, 64), qb1, s1);

    // lane l holds query (l&15); vgpr j -> key j+8*hh (s0) / 16+j+8*hh (s1)
    float cmax = -1e30f;
#pragma unroll
    for (int j = 0; j < 8; ++j) {
      s0[j] *= scale;
      s1[j] *= scale;
      cmax = fmaxf(cmax, fmaxf(s0[j], s1[j]));
    }
    cmax = fmaxf(cmax, __shfl_xor(cmax, 16, 32));  // combine lane pair
    const float mn = fmaxf(mrun, cmax);
    const float cr = __expf(mrun - mn);
    mrun = mn;

    float rs = 0.f;
    v8h pk0, pk1;
#pragma unroll
    for (int j = 0; j < 8; ++j) {
      const float p0 = __expf(s0[j] - mn);
      const float p1 = __expf(s1[j] - mn);
      rs += p0 + p1;
      pk0[j] = (_Float16)p0;
      pk1[j] = (_Float16)p1;
    }
    // P row-major [query][key]: two contiguous b128 stores per lane
    *(v8h*)(Pw + lane_q * 32 + 8 * hh)      = pk0;
    *(v8h*)(Pw + lane_q * 32 + 16 + 8 * hh) = pk1;
    rs += __shfl_xor(rs, 16, 32);
    lrun = lrun * cr + rs;

#pragma unroll
    for (int nt = 0; nt < 4; ++nt)
#pragma unroll
      for (int j = 0; j < 8; ++j) O[nt][j] *= cr;  // per-lane scalar rescale

    // O^T += V^T x P^T : both fragments contiguous b128 loads now
    const v16h bP = frag_row16(Pw, 32);
#pragma unroll
    for (int nt = 0; nt < 4; ++nt) {
      const v16h aV = frag_row16(Vc + (nt * 16) * 32, 32);
      O[nt] = wmma_f16(aV, bP, O[nt]);
    }

    if (pref) {
      *(v8h*)(&Vts[p ^ 1][vdl * 32 + vks]) = rv;  // publish next V
      wait_stage();                               // my K async landed
    }
    __syncthreads();
  }

  // O^T: lane = query, vgpr (nt,j) -> d = nt*16 + j + 8*hh. Normalize and
  // transpose through per-wave LDS tile (same-wave DS ops are in-order).
  const float inv = outw / lrun;
#pragma unroll
  for (int nt = 0; nt < 4; ++nt) {
    float4 w0, w1;
    w0.x = O[nt][0] * inv; w0.y = O[nt][1] * inv;
    w0.z = O[nt][2] * inv; w0.w = O[nt][3] * inv;
    w1.x = O[nt][4] * inv; w1.y = O[nt][5] * inv;
    w1.z = O[nt][6] * inv; w1.w = O[nt][7] * inv;
    float* dst = Ow + lane_q * 64 + nt * 16 + 8 * hh;
    *(float4*)(dst) = w0;
    *(float4*)(dst + 4) = w1;
  }

  // coalesced read-back: lane l -> row l>>1, 32-float half (l&1)
  const int r = l >> 1, dp = (l & 1) * 32;
  float* cdst = ctx + ((size_t)batch * Lq + qrow0 + r) * E_DIM + h * HD + dp;
  const float* osrc = Ow + r * 64 + dp;
#pragma unroll
  for (int i = 0; i < 8; ++i) {
    float4 v = *(const float4*)(osrc + i * 4);
    if (accumulate) {
      const float4 c = *(const float4*)(cdst + i * 4);
      v.x += c.x; v.y += c.y; v.z += c.z; v.w += c.w;
    }
    *(float4*)(cdst + i * 4) = v;
  }
}

// ---------------------------------------------------------------------------
extern "C" void kernel_launch(void* const* d_in, const int* in_sizes, int n_in,
                              void* d_out, int out_size, void* d_ws,
                              size_t ws_size, hipStream_t stream) {
  (void)in_sizes; (void)n_in; (void)out_size; (void)ws_size;
  const float* query = (const float*)d_in[0];
  const float* key   = (const float*)d_in[1];
  const float* value = (const float*)d_in[2];
  const float* Wq = (const float*)d_in[3];
  const float* bq = (const float*)d_in[4];
  const float* Wk = (const float*)d_in[5];
  const float* bk = (const float*)d_in[6];
  const float* Wv = (const float*)d_in[7];
  const float* bv = (const float*)d_in[8];
  const float* Wo = (const float*)d_in[9];
  const float* bo = (const float*)d_in[10];
  const float* smask = (const float*)d_in[11];
  float* out = (float*)d_out;

  char* ws = (char*)d_ws;
  const size_t SZ = (size_t)32 * 512 * 1024 * 2;  // 32MB: full-length f16 tensor
  const size_t WSZ = (size_t)E_DIM * E_DIM * 2;   // 2MB: f16 transposed weight
  _Float16* Qp  = (_Float16*)(ws);
  _Float16* Kp  = (_Float16*)(ws + SZ);
  _Float16* Vp  = (_Float16*)(ws + 2 * SZ);
  _Float16* Kd1 = (_Float16*)(ws + 3 * SZ);            // 16MB
  _Float16* Vd1 = (_Float16*)(ws + 3 * SZ + SZ / 2);   // 16MB
  _Float16* Kd2 = (_Float16*)(ws + 4 * SZ);            // 8MB
  _Float16* Vd2 = (_Float16*)(ws + 4 * SZ + SZ / 4);   // 8MB
  float* ctx    = (float*)(ws + 4 * SZ + SZ / 2);      // 64MB f32
  char* wbase   = ws + 4 * SZ + SZ / 2 + (size_t)32 * 512 * E_DIM * 4;
  _Float16* WqT = (_Float16*)(wbase);
  _Float16* WkT = (_Float16*)(wbase + WSZ);
  _Float16* WvT = (_Float16*)(wbase + 2 * WSZ);
  _Float16* WoT = (_Float16*)(wbase + 3 * WSZ);

  const int M = 32 * 512, N = E_DIM, K = E_DIM;
  const dim3 gg(N / 128, M / 128), bb(256);
  const dim3 gt(N / 32, K / 32);

  // One-time weight transposes (f32 KxN -> f16 NxK) for async B staging
  transpose_w_kernel<<<gt, bb, 0, stream>>>(Wq, WqT, N, K);
  transpose_w_kernel<<<gt, bb, 0, stream>>>(Wk, WkT, N, K);
  transpose_w_kernel<<<gt, bb, 0, stream>>>(Wv, WvT, N, K);
  transpose_w_kernel<<<gt, bb, 0, stream>>>(Wo, WoT, N, K);

  // Shared QKV projections (downsampling commutes with the linear projection)
  wmma_gemm_kernel<0><<<gg, bb, 0, stream>>>(query, WqT, bq, 1.0f, Qp, nullptr,
                                             nullptr, M, N, K, 512);
  wmma_gemm_kernel<0><<<gg, bb, 0, stream>>>(key, WkT, bk, 1.0f, Kp, nullptr,
                                             nullptr, M, N, K, 512);
  wmma_gemm_kernel<0><<<gg, bb, 0, stream>>>(value, WvT, bv, 1.0f, Vp, nullptr,
                                             nullptr, M, N, K, 512);

  {  // hierarchical downsampling of the *projected* K/V
    const size_t tot1 = (size_t)32 * 256 * E_DIM;
    const int nb1 = (int)((tot1 + 255) / 256);
    downsample2_kernel<<<nb1, 256, 0, stream>>>(Kp, Kd1, 256);
    downsample2_kernel<<<nb1, 256, 0, stream>>>(Vp, Vd1, 256);
    const size_t tot2 = (size_t)32 * 128 * E_DIM;
    const int nb2 = (int)((tot2 + 255) / 256);
    downsample2_kernel<<<nb2, 256, 0, stream>>>(Kd1, Kd2, 128);
    downsample2_kernel<<<nb2, 256, 0, stream>>>(Vd1, Vd2, 128);
  }

  // 3 hierarchy levels (weight 1/3) + sliding window (weight 1), all
  // accumulated into ctx; the first launch initializes ctx (poisoned ws).
  const float w3 = 1.0f / 3.0f;
  flash_attn_kernel<<<dim3(32 * HEADS, 4), bb, 0, stream>>>(Qp, Kp, Vp, ctx,
                                                            512, 512, w3, 0);
  flash_attn_kernel<<<dim3(32 * HEADS, 4), bb, 0, stream>>>(Qp, Kd1, Vd1, ctx,
                                                            512, 256, w3, 1);
  flash_attn_kernel<<<dim3(32 * HEADS, 4), bb, 0, stream>>>(Qp, Kd2, Vd2, ctx,
                                                            512, 128, w3, 1);
  // sliding window == block-diagonal: reshape (B,512,E) -> (B*4,128,E) is a view
  flash_attn_kernel<<<dim3(128 * HEADS, 1), bb, 0, stream>>>(Qp, Kp, Vp, ctx,
                                                             128, 128, 1.0f, 1);

  // Single folded output projection: (ctx)@Wo + 2*bo, gated by sigmoid(mask)
  wmma_gemm_kernel<1><<<gg, bb, 0, stream>>>(ctx, WoT, bo, 2.0f, nullptr, out,
                                             smask, M, N, K, 512);
}
